// MMD_29729763623087
// MI455X (gfx1250) — matrix operand order repaired and checked by reading, not compiled, not run on hardware
//
#include <hip/hip_runtime.h>

typedef __attribute__((ext_vector_type(16))) _Float16 v16h;
typedef __attribute__((ext_vector_type(8)))  _Float16 v8h;
typedef __attribute__((ext_vector_type(4)))  _Float16 v4h;
typedef __attribute__((ext_vector_type(8)))  float    v8f;
typedef __attribute__((ext_vector_type(4)))  float    v4f;

#define B_    16
#define NWAY  5
#define NS    130
#define NSP   160   // padded to 10 tiles of 16 -> 5x5 superblocks of 32x32
#define Q_    75
#define NQ    25
#define NQP   32    // padded to 2 tiles of 16 -> 1 superblock
#define C_    640
#define KSTEP 32
#define KITER (C_ / KSTEP)   // 20

// alphas = 2^-3..2^1 -> sum_a exp(-a d2) = e + e^2 + e^4 + e^8 + e^16, e = exp(-d2/8)
__device__ __forceinline__ float multi_gauss(float d2) {
    float e   = __expf(-0.125f * d2);
    float e2  = e * e;
    float e4  = e2 * e2;
    float e8  = e4 * e4;
    float e16 = e8 * e8;
    return e + e2 + e4 + e8 + e16;
}

__device__ __forceinline__ float wave_reduce(float v) {
    #pragma unroll
    for (int off = 16; off > 0; off >>= 1)
        v += __shfl_xor(v, off, 32);
    return v;
}

// A 16x32 f16 fragment at p (kb pre-folded): halves {0..7} and {16..23}
__device__ __forceinline__ v16h ld_frag_a(const _Float16* p) {
    v8h lo = *(const v8h*)p;
    v8h hi = *(const v8h*)(p + 16);
    v16h a;
    #pragma unroll
    for (int i = 0; i < 8; ++i) { a[i] = lo[i]; a[8 + i] = hi[i]; }
    return a;
}
// B 32x16 f16 fragment at p (kb pre-folded): halves {0..7} and {16..23}
__device__ __forceinline__ v16h ld_frag_b(const _Float16* p) {
    v8h lo = *(const v8h*)p;
    v8h hi = *(const v8h*)(p + 16);
    v16h b;
    #pragma unroll
    for (int i = 0; i < 8; ++i) { b[i] = lo[i]; b[8 + i] = hi[i]; }
    return b;
}

// branchless masked multi-gauss sum over one 16x16 C tile
// mbase = tile_m0 + 8*(lane>=16), ncol = tile_n0 + (lane&15)
__device__ __forceinline__ float tile_sum(const v8f& c, int mbase, int ncol,
                                          int mv, int nv, bool excl_diag) {
    float s = 0.f;
    #pragma unroll
    for (int r = 0; r < 8; ++r) {
        int m = mbase + r;
        float d2 = fmaxf(2.0f - 2.0f * c[r], 0.0f);
        float g  = multi_gauss(d2);
        bool ok = (m < mv) & (ncol < nv) & (!excl_diag | (m != ncol));
        s += ok ? g : 0.0f;
    }
    return s;
}

// 32x32 superblock: 2x2 WMMA tiles with shared A/B fragments (4 loads/wmma)
// returns this lane's masked kernel-sum contribution
__device__ __forceinline__ float super_block(const _Float16* slabA, const _Float16* slabB,
                                             int m0, int n0, int lane,
                                             int mv, int nv, bool excl_diag) {
    int mo = (lane & 16) ? 8 : 0;       // C-layout row offset within tile
    int nc = lane & 15;                 // C-layout column within tile
    const _Float16* pa0 = slabA + (size_t)(m0 + (lane & 15)) * C_ + ((lane & 16) ? 8 : 0);
    const _Float16* pa1 = pa0 + 16 * C_;
    const _Float16* pb0 = slabB + (size_t)(n0 + (lane & 15)) * C_ + ((lane & 16) ? 8 : 0);
    const _Float16* pb1 = pb0 + 16 * C_;
    v8f c00 = {}, c01 = {}, c10 = {}, c11 = {};
    #pragma unroll 2
    for (int k = 0; k < KITER; ++k) {
        v16h a0 = ld_frag_a(pa0);
        v16h a1 = ld_frag_a(pa1);
        v16h b0 = ld_frag_b(pb0);
        v16h b1 = ld_frag_b(pb1);
        c00 = __builtin_amdgcn_wmma_f32_16x16x32_f16(false, a0, false, b0, (short)0, c00, false, false);
        c01 = __builtin_amdgcn_wmma_f32_16x16x32_f16(false, a0, false, b1, (short)0, c01, false, false);
        c10 = __builtin_amdgcn_wmma_f32_16x16x32_f16(false, a1, false, b0, (short)0, c10, false, false);
        c11 = __builtin_amdgcn_wmma_f32_16x16x32_f16(false, a1, false, b1, (short)0, c11, false, false);
        pa0 += KSTEP; pa1 += KSTEP; pb0 += KSTEP; pb1 += KSTEP;
    }
    float s = 0.f;
    s += tile_sum(c00, m0      + mo, n0      + nc, mv, nv, excl_diag);
    s += tile_sum(c01, m0      + mo, n0 + 16 + nc, mv, nv, excl_diag);
    s += tile_sum(c10, m0 + 16 + mo, n0      + nc, mv, nv, excl_diag);
    s += tile_sum(c11, m0 + 16 + mo, n0 + 16 + nc, mv, nv, excl_diag);
    return s;
}

// --- preprocess: center over C, l2-normalize, emit f16 into padded slab ---
// one wave32 per padded row; pad rows get zeros
__global__ void mmd_prep_kernel(const float* __restrict__ src,
                                _Float16* __restrict__ dst,
                                int rows_valid, int rows_pad) {
    int wave = threadIdx.x >> 5;
    int lane = threadIdx.x & 31;
    int pr   = blockIdx.x * 8 + wave;            // padded row id
    int slab = pr / rows_pad;
    int prow = pr - slab * rows_pad;
    _Float16* drow = dst + (size_t)pr * C_;
    if (prow >= rows_valid) {
        v4h z = {};
        #pragma unroll
        for (int i = 0; i < 5; ++i)
            *(v4h*)(drow + lane * 4 + i * 128) = z;
        return;
    }
    const float* srow = src + ((size_t)slab * rows_valid + prow) * C_;
    float v[20];
    float s = 0.f;
    #pragma unroll
    for (int i = 0; i < 5; ++i) {
        v4f x = *(const v4f*)(srow + lane * 4 + i * 128);
        #pragma unroll
        for (int j = 0; j < 4; ++j) { v[4 * i + j] = x[j]; s += x[j]; }
    }
    s = wave_reduce(s);
    float mean = s * (1.0f / (float)C_);
    float ss = 0.f;
    #pragma unroll
    for (int i = 0; i < 20; ++i) { v[i] -= mean; ss += v[i] * v[i]; }
    ss = wave_reduce(ss);
    float scale = rsqrtf(ss + 1e-12f);
    #pragma unroll
    for (int i = 0; i < 5; ++i) {
        v4h h;
        #pragma unroll
        for (int j = 0; j < 4; ++j) h[j] = (_Float16)(v[4 * i + j] * scale);
        *(v4h*)(drow + lane * 4 + i * 128) = h;
    }
}

// --- self MMD: sum_{i!=j, i,j<n_valid} multi_gauss(2 - 2*gram) / (n(n-1)) ---
__global__ void mmd_self_kernel(const _Float16* __restrict__ Xh,
                                float* __restrict__ out,
                                int n_valid, int n_pad) {
    const _Float16* base = Xh + (size_t)blockIdx.x * n_pad * C_;
    int lane   = threadIdx.x & 31;
    int wave   = threadIdx.x >> 5;
    int nwaves = blockDim.x >> 5;
    int sd  = n_pad >> 5;                // superblocks per dim
    int nsb = sd * sd;
    float acc = 0.f;
    for (int t = wave; t < nsb; t += nwaves) {
        int m0 = (t / sd) * 32;
        int n0 = (t % sd) * 32;
        acc += super_block(base, base, m0, n0, lane, n_valid, n_valid, true);
    }
    __shared__ float red[8];
    acc = wave_reduce(acc);
    if (lane == 0) red[wave] = acc;
    __syncthreads();
    if (threadIdx.x == 0) {
        float tot = 0.f;
        for (int w = 0; w < nwaves; ++w) tot += red[w];
        out[blockIdx.x] = tot / (float)(n_valid * (n_valid - 1));
    }
}

// --- cross MMD + combine: out[b,q,w] = mmd_s + mmd_q - 2 * sum(k_qs)/(ns*nq) ---
// blockDim = 160 (5 waves); wave w handles support superblock n0 = 32*w
__global__ void mmd_cross_kernel(const _Float16* __restrict__ Sh,
                                 const _Float16* __restrict__ Th,
                                 const float* __restrict__ mmd_s,
                                 const float* __restrict__ mmd_q,
                                 float* __restrict__ out) {
    int w = blockIdx.x;   // n_way
    int q = blockIdx.y;
    int b = blockIdx.z;
    const _Float16* sbase = Sh + (size_t)(b * NWAY + w) * NSP * C_;
    const _Float16* tbase = Th + (size_t)(b * Q_ + q) * NQP * C_;
    int lane   = threadIdx.x & 31;
    int wave   = threadIdx.x >> 5;
    int nwaves = blockDim.x >> 5;
    float acc = 0.f;
    for (int t = wave; t < NSP / 32; t += nwaves) {
        // M = query rows (32 padded), N = support rows (superblock t)
        acc += super_block(tbase, sbase, 0, t * 32, lane, NQ, NS, false);
    }
    __shared__ float red[8];
    acc = wave_reduce(acc);
    if (lane == 0) red[wave] = acc;
    __syncthreads();
    if (threadIdx.x == 0) {
        float tot = 0.f;
        for (int i = 0; i < nwaves; ++i) tot += red[i];
        float mqs = tot / (float)(NS * NQ);
        out[((size_t)b * Q_ + q) * NWAY + w] =
            mmd_s[b * NWAY + w] + mmd_q[b * Q_ + q] - 2.0f * mqs;
    }
}

extern "C" void kernel_launch(void* const* d_in, const int* in_sizes, int n_in,
                              void* d_out, int out_size, void* d_ws, size_t ws_size,
                              hipStream_t stream) {
    (void)in_sizes; (void)n_in; (void)out_size; (void)ws_size;
    const float* support = (const float*)d_in[0];  // [16,5,130,640] f32
    const float* query   = (const float*)d_in[1];  // [16,75,25,640] f32
    float* out = (float*)d_out;                    // [16,75,5] f32

    char* ws = (char*)d_ws;
    _Float16* S_h = (_Float16*)ws;                                 // [16*5,160,640] f16
    size_t s_bytes = (size_t)B_ * NWAY * NSP * C_ * sizeof(_Float16);
    _Float16* T_h = (_Float16*)(ws + s_bytes);                     // [16*75,32,640] f16
    size_t t_bytes = (size_t)B_ * Q_ * NQP * C_ * sizeof(_Float16);
    float* mmd_s = (float*)(ws + s_bytes + t_bytes);               // [16*5]
    float* mmd_q = mmd_s + B_ * NWAY;                              // [16*75]

    // center + normalize + f16 pack (8 rows per 256-thread block)
    mmd_prep_kernel<<<(B_ * NWAY * NSP) / 8, 256, 0, stream>>>(support, S_h, NS, NSP);
    mmd_prep_kernel<<<(B_ * Q_ * NQP) / 8, 256, 0, stream>>>(query, T_h, NQ, NQP);

    // unbiased self terms
    mmd_self_kernel<<<B_ * NWAY, 256, 0, stream>>>(S_h, mmd_s, NS, NSP);  // 5x5 superblocks, 8 waves
    mmd_self_kernel<<<B_ * Q_, 32, 0, stream>>>(T_h, mmd_q, NQ, NQP);     // 1 superblock, 1 wave

    // cross term + combine (5 support superblocks -> 5 waves)
    dim3 grid(NWAY, Q_, B_);
    mmd_cross_kernel<<<grid, 160, 0, stream>>>(S_h, T_h, mmd_s, mmd_q, out);
}
